// Net_10015863734378
// MI455X (gfx1250) — matrix operand order repaired and checked by reading, not compiled, not run on hardware
//
#include <hip/hip_runtime.h>
#include <hip/hip_bf16.h>

#define NN    100000
#define EE    3200000
#define FIN   512
#define HID   64
#define NC    40
#define NBINS 65536
#define GCAP  4096
#define RANK_K 1600000   // E - int(E*0.5): index into ascending sort

typedef __attribute__((ext_vector_type(16))) __bf16 v16bf;
typedef __attribute__((ext_vector_type(8)))  float  v8f;

#if __has_builtin(__builtin_amdgcn_global_load_async_to_lds_b128) && \
    __has_builtin(__builtin_amdgcn_s_wait_asynccnt)
#define USE_ASYNC_LDS 1
// Match the builtin's exact parameter types (from hipcc diagnostics):
//   arg0: int __attribute__((vector_size(16))) __device__ *   (AS1 global)
//   arg1: same pointee in AS3 (LDS)
typedef int v4i_vs __attribute__((vector_size(16)));
typedef __attribute__((address_space(1))) v4i_vs* gld_gptr;
typedef __attribute__((address_space(3))) v4i_vs* gld_lptr;

template <int OFF>
__device__ __forceinline__ void async_copy16(const void* g, void* l) {
    __builtin_amdgcn_global_load_async_to_lds_b128((gld_gptr)g, (gld_lptr)l, OFF, 0);
}
#else
#define USE_ASYNC_LDS 0
#endif

// ---------------- utility ----------------
__global__ void zero_u32_kernel(unsigned* p, long long n) {
    long long i = (long long)blockIdx.x * 256 + threadIdx.x;
    if (i < n) p[i] = 0u;
}

// ---------------- degree / norm ----------------
__global__ void deg_kernel(const int* __restrict__ dst, float* __restrict__ deg) {
    int e = blockIdx.x * 256 + threadIdx.x;
    if (e < EE) unsafeAtomicAdd(&deg[dst[e]], 1.0f);
}

__global__ void dis_kernel(float* degdis) {
    int i = blockIdx.x * 256 + threadIdx.x;
    if (i < NN) degdis[i] = rsqrtf(degdis[i] + 1.0f);
}

// ---------------- weight conversion into WMMA-fragment-swizzled bf16 ----------------
// w1s[((kt*4+nt)*32+lane)*16+e] = w1[K*HID+N], K=kt*32+(lane>>4)*8+(e&7)+(e>>3)*16, N=nt*16+(lane&15)
__global__ void conv_w1_kernel(const float* __restrict__ w1, __bf16* __restrict__ w1s) {
    int i = blockIdx.x * 256 + threadIdx.x;
    if (i < FIN * HID) {
        int e    = i & 15;
        int lane = (i >> 4) & 31;
        int rest = i >> 9;
        int nt   = rest & 3;
        int kt   = rest >> 2;
        int K = kt * 32 + (lane >> 4) * 8 + (e & 7) + (e >> 3) * 16;
        int N = nt * 16 + (lane & 15);
        w1s[i] = (__bf16)w1[K * HID + N];
    }
}

// w2s[((kt*3+nt)*32+lane)*16+e]; K=kt*32+(lane>>4)*8+(e&7)+(e>>3)*16, N=nt*16+(lane&15); pad N>=40 with 0
__global__ void conv_w2_kernel(const float* __restrict__ w2, __bf16* __restrict__ w2s) {
    int i = blockIdx.x * 256 + threadIdx.x;
    if (i < 2 * 3 * 32 * 16) {
        int e    = i & 15;
        int lane = (i >> 4) & 31;
        int rest = i >> 9;
        int nt   = rest % 3;
        int kt   = rest / 3;
        int K = kt * 32 + (lane >> 4) * 8 + (e & 7) + (e >> 3) * 16;
        int N = nt * 16 + (lane & 15);
        w2s[i] = (N < NC) ? (__bf16)w2[K * NC + N] : (__bf16)0.0f;
    }
}

__device__ __forceinline__ v16bf cvt_frag(float4 a0, float4 a1, float4 a2, float4 a3) {
    v16bf a;
    a[0]=(__bf16)a0.x; a[1]=(__bf16)a0.y; a[2]=(__bf16)a0.z; a[3]=(__bf16)a0.w;
    a[4]=(__bf16)a1.x; a[5]=(__bf16)a1.y; a[6]=(__bf16)a1.z; a[7]=(__bf16)a1.w;
    a[8]=(__bf16)a2.x; a[9]=(__bf16)a2.y; a[10]=(__bf16)a2.z; a[11]=(__bf16)a2.w;
    a[12]=(__bf16)a3.x; a[13]=(__bf16)a3.y; a[14]=(__bf16)a3.z; a[15]=(__bf16)a3.w;
    return a;
}

// ---------------- GEMM1: h0[N,64] = x[N,512] @ w1[512,64] ----------------
// One wave per 16-row M tile; each wave computes all 4 N tiles so x is read once.
__global__ __launch_bounds__(64) void gemm1_kernel(const float* __restrict__ x,
                                                   const __bf16* __restrict__ w1s,
                                                   float* __restrict__ h0) {
    const int wv    = threadIdx.x >> 5;
    const int lane  = threadIdx.x & 31;
    const int half  = lane >> 4;
    const int l15   = lane & 15;
    const int mtile = blockIdx.x * 2 + wv;       // 3125 blocks * 2 waves = 6250 tiles
    const int row   = mtile * 16 + l15;

    v8f acc[4] = {{}, {}, {}, {}};

#if USE_ASYNC_LDS
    // Double-buffered A-tile staging: 16 rows x 32 floats = 2KB per buffer per wave.
    __shared__ float sbuf[2][2][512];
    const int seg = half;                        // lane stages row (lane&15), 16-float segment
    const float* gsrc_row = x + row * FIN + seg * 16;

    {   // stage kt=0 into buffer 0
        void* ld = (void*)&sbuf[wv][0][l15 * 32 + seg * 16];
        const void* gs = (const void*)gsrc_row;
        async_copy16<0>(gs, ld);
        async_copy16<16>(gs, ld);
        async_copy16<32>(gs, ld);
        async_copy16<48>(gs, ld);
    }
    asm volatile("" ::: "memory");
    for (int kt = 0; kt < FIN / 32; ++kt) {
        if (kt + 1 < FIN / 32) {
            void* ld = (void*)&sbuf[wv][(kt + 1) & 1][l15 * 32 + seg * 16];
            const void* gs = (const void*)(gsrc_row + (kt + 1) * 32);
            async_copy16<0>(gs, ld);
            async_copy16<16>(gs, ld);
            async_copy16<32>(gs, ld);
            async_copy16<48>(gs, ld);
            asm volatile("" ::: "memory");
            __builtin_amdgcn_s_wait_asynccnt(4);   // current buffer's 4 copies done
        } else {
            __builtin_amdgcn_s_wait_asynccnt(0);
        }
        asm volatile("" ::: "memory");
        const float4* lv = (const float4*)sbuf[wv][kt & 1];
        int idx4 = l15 * 8 + half * 2;
        v16bf a = cvt_frag(lv[idx4], lv[idx4 + 1], lv[idx4 + 4], lv[idx4 + 5]);
        #pragma unroll
        for (int nt = 0; nt < 4; ++nt) {
            v16bf b = *(const v16bf*)&w1s[((kt * 4 + nt) * 32 + lane) * 16];
            acc[nt] = __builtin_amdgcn_wmma_f32_16x16x32_bf16(false, a, false, b,
                                                              (short)0, acc[nt], false, false);
        }
    }
#else
    const float4* xv = (const float4*)x;
    for (int kt = 0; kt < FIN / 32; ++kt) {
        int idx4 = row * (FIN / 4) + kt * 8 + half * 2;
        float4 a0 = xv[idx4];
        float4 a1 = xv[idx4 + 1];
        float4 a2 = xv[idx4 + 4];
        float4 a3 = xv[idx4 + 5];
        __builtin_prefetch(&x[row * FIN + (kt + 1) * 32 + half * 8], 0, 1);
        v16bf a = cvt_frag(a0, a1, a2, a3);
        #pragma unroll
        for (int nt = 0; nt < 4; ++nt) {
            v16bf b = *(const v16bf*)&w1s[((kt * 4 + nt) * 32 + lane) * 16];
            acc[nt] = __builtin_amdgcn_wmma_f32_16x16x32_bf16(false, a, false, b,
                                                              (short)0, acc[nt], false, false);
        }
    }
#endif

    // C/D: VGPR e -> M = half*8 + e, N = nt*16 + (lane&15)
    int m0 = mtile * 16 + half * 8;
    #pragma unroll
    for (int nt = 0; nt < 4; ++nt) {
        int col = nt * 16 + l15;
        #pragma unroll
        for (int e = 0; e < 8; ++e) h0[(m0 + e) * HID + col] = acc[nt][e];
    }
}

// ---------------- GEMM2: g2[N,40] = h1[N,64] @ w2[64,40] (N padded to 48) ----------------
__global__ __launch_bounds__(64) void gemm2_kernel(const float* __restrict__ h1,
                                                   const __bf16* __restrict__ w2s,
                                                   float* __restrict__ g2) {
    const int wv    = threadIdx.x >> 5;
    const int lane  = threadIdx.x & 31;
    const int half  = lane >> 4;
    const int l15   = lane & 15;
    const int mtile = blockIdx.x * 2 + wv;
    const int row   = mtile * 16 + l15;
    const float4* hv = (const float4*)h1;

    v8f acc[3] = {{}, {}, {}};
    #pragma unroll
    for (int kt = 0; kt < HID / 32; ++kt) {
        int idx4 = row * (HID / 4) + kt * 8 + half * 2;
        float4 a0 = hv[idx4];
        float4 a1 = hv[idx4 + 1];
        float4 a2 = hv[idx4 + 4];
        float4 a3 = hv[idx4 + 5];
        v16bf a = cvt_frag(a0, a1, a2, a3);
        #pragma unroll
        for (int nt = 0; nt < 3; ++nt) {
            v16bf b = *(const v16bf*)&w2s[((kt * 3 + nt) * 32 + lane) * 16];
            acc[nt] = __builtin_amdgcn_wmma_f32_16x16x32_bf16(false, a, false, b,
                                                              (short)0, acc[nt], false, false);
        }
    }
    int m0 = mtile * 16 + half * 8;
    #pragma unroll
    for (int nt = 0; nt < 3; ++nt) {
        int col = nt * 16 + l15;
        if (col < NC) {
            #pragma unroll
            for (int e = 0; e < 8; ++e) g2[(m0 + e) * NC + col] = acc[nt][e];
        }
    }
}

// ---------------- edge propagation (scatter-add) ----------------
__global__ __launch_bounds__(256) void prop1_kernel(const int* __restrict__ src,
                                                    const int* __restrict__ dst,
                                                    const float* __restrict__ dis,
                                                    const float* __restrict__ h0,
                                                    float* __restrict__ agg) {
    long long w = (long long)blockIdx.x * 8 + (threadIdx.x >> 5);  // wave == edge
    int lane = threadIdx.x & 31;
    if (w < EE) {
        int s = src[w], d = dst[w];
        float nrm = dis[s] * dis[d];
        float v0 = h0[s * HID + lane] * nrm;
        float v1 = h0[s * HID + lane + 32] * nrm;
        unsafeAtomicAdd(&agg[d * HID + lane], v0);
        unsafeAtomicAdd(&agg[d * HID + lane + 32], v1);
    }
}

__global__ __launch_bounds__(256) void prop2_kernel(const int* __restrict__ src,
                                                    const int* __restrict__ dst,
                                                    const float* __restrict__ dis,
                                                    const float* __restrict__ g2,
                                                    float* __restrict__ agg) {
    long long w = (long long)blockIdx.x * 8 + (threadIdx.x >> 5);
    int lane = threadIdx.x & 31;
    if (w < EE) {
        int s = src[w], d = dst[w];
        float nrm = dis[s] * dis[d];
        unsafeAtomicAdd(&agg[d * NC + lane], g2[s * NC + lane] * nrm);
        if (lane < NC - 32)
            unsafeAtomicAdd(&agg[d * NC + 32 + lane], g2[s * NC + 32 + lane] * nrm);
    }
}

// ---------------- self-loop + bias (+relu for layer1); in-place into agg ----------------
__global__ void selfloop1_kernel(const float* __restrict__ h0, float* __restrict__ agg,
                                 const float* __restrict__ dis, const float* __restrict__ b1) {
    int idx = blockIdx.x * 256 + threadIdx.x;
    if (idx < NN * HID) {
        int i = idx / HID, f = idx % HID;
        float di = dis[i];
        agg[idx] = fmaxf(agg[idx] + h0[idx] * di * di + b1[f], 0.0f);
    }
}

__global__ void selfloop2_kernel(const float* __restrict__ g2, float* __restrict__ agg,
                                 const float* __restrict__ dis, const float* __restrict__ b2) {
    int idx = blockIdx.x * 256 + threadIdx.x;
    if (idx < NN * NC) {
        int i = idx / NC, c = idx % NC;
        float di = dis[i];
        agg[idx] = agg[idx] + g2[idx] * di * di + b2[c];
    }
}

// ---------------- exact k-th order statistic of edge_weight ----------------
__device__ __forceinline__ int weight_bin(float w) {
    int b = (int)(w * 65536.0f);
    return b < 0 ? 0 : (b > NBINS - 1 ? NBINS - 1 : b);
}

__global__ void hist_kernel(const float* __restrict__ ew, int* __restrict__ hist) {
    int e = blockIdx.x * 256 + threadIdx.x;
    if (e < EE) atomicAdd(&hist[weight_bin(ew[e])], 1);
}

__global__ void scan_select_kernel(const int* __restrict__ hist, int* __restrict__ misc) {
    if (blockIdx.x == 0 && threadIdx.x == 0) {
        long long cum = 0; long long k = RANK_K;
        for (int b = 0; b < NBINS; ++b) {
            int c = hist[b];
            if (cum + c > k) { misc[1] = b; misc[2] = (int)(k - cum); return; }
            cum += c;
        }
        misc[1] = NBINS - 1; misc[2] = 0;
    }
}

__global__ void gather_kernel(const float* __restrict__ ew, int* __restrict__ misc,
                              float* __restrict__ buf) {
    int e = blockIdx.x * 256 + threadIdx.x;
    if (e < EE) {
        float w = ew[e];
        if (weight_bin(w) == misc[1]) {
            int p = atomicAdd(&misc[0], 1);
            if (p < GCAP) buf[p] = w;
        }
    }
}

__global__ void final_select_kernel(int* __restrict__ misc, float* __restrict__ buf) {
    if (blockIdx.x == 0 && threadIdx.x == 0) {
        float* miscf = (float*)misc;
        int n = misc[0]; if (n > GCAP) n = GCAP;
        int r = misc[2];
        float th = 1.0f;
        for (int it = 0; it <= r && it < n; ++it) {   // r-th smallest in bin
            float mn = 3.4e38f; int mi = -1;
            for (int j = 0; j < n; ++j) if (buf[j] < mn) { mn = buf[j]; mi = j; }
            th = mn;
            if (mi >= 0) buf[mi] = 3.4e38f;
        }
        miscf[3] = th;
    }
}

// ---------------- edge loss ----------------
__global__ __launch_bounds__(256) void loss_kernel(const int* __restrict__ src,
                                                   const int* __restrict__ dst,
                                                   const float* __restrict__ x2,
                                                   const float* __restrict__ ew,
                                                   const float* __restrict__ lu,
                                                   const float* __restrict__ miscf,
                                                   float* __restrict__ accum) {
    __shared__ float red[256];
    int e = blockIdx.x * 256 + threadIdx.x;
    float c = 0.0f;
    if (e < EE) {
        int s = src[e], d = dst[e];
        const float* a = &x2[s * NC];
        const float* b = &x2[d * NC];
        float dot = 0.f, na = 0.f, nb = 0.f;
        #pragma unroll
        for (int j = 0; j < NC; ++j) {
            float av = a[j], bv = b[j];
            dot += av * bv; na += av * av; nb += bv * bv;
        }
        na = fmaxf(sqrtf(na), 1e-8f);
        nb = fmaxf(sqrtf(nb), 1e-8f);
        float cs = 1.0f - dot / (na * nb);
        float w = ew[e];
        bool m = (w >= miscf[3]);
        float lp = m ? cs : 1.0f - cs;
        float le = m ? w : 1.0f - w;
        c = le * lp * lu[e];
    }
    red[threadIdx.x] = c;
    __syncthreads();
    for (int s2 = 128; s2 > 0; s2 >>= 1) {
        if (threadIdx.x < s2) red[threadIdx.x] += red[threadIdx.x + s2];
        __syncthreads();
    }
    if (threadIdx.x == 0) unsafeAtomicAdd(accum, red[0]);
}

// ---------------- log_softmax + final scalar ----------------
__global__ void logsoftmax_kernel(const float* __restrict__ x2, float* __restrict__ out) {
    int i = blockIdx.x * 256 + threadIdx.x;
    if (i < NN) {
        const float* r = &x2[i * NC];
        float m = -3.4e38f;
        #pragma unroll
        for (int c = 0; c < NC; ++c) m = fmaxf(m, r[c]);
        float s = 0.f;
        #pragma unroll
        for (int c = 0; c < NC; ++c) s += expf(r[c] - m);
        float l = logf(s);
        #pragma unroll
        for (int c = 0; c < NC; ++c) out[i * NC + c] = r[c] - m - l;
    }
}

__global__ void finalize_kernel(const float* __restrict__ miscf, float* __restrict__ out) {
    if (blockIdx.x == 0 && threadIdx.x == 0)
        out[NN * NC] = miscf[4] / (float)EE;
}

// ---------------- host launcher ----------------
extern "C" void kernel_launch(void* const* d_in, const int* in_sizes, int n_in,
                              void* d_out, int out_size, void* d_ws, size_t ws_size,
                              hipStream_t stream) {
    const float* x  = (const float*)d_in[0];
    const int*   ei = (const int*)d_in[1];
    const float* ew = (const float*)d_in[2];
    const float* lu = (const float*)d_in[3];
    const float* w1 = (const float*)d_in[4];
    const float* b1 = (const float*)d_in[5];
    const float* w2 = (const float*)d_in[6];
    const float* b2 = (const float*)d_in[7];
    const int* src = ei;
    const int* dst = ei + EE;

    char* ws = (char*)d_ws;
    float*  h0   = (float*)(ws);                 // 25.6 MB
    float*  agg1 = (float*)(ws + 25600000);      // 25.6 MB (becomes h1 in-place)
    float*  g2   = (float*)(ws + 51200000);      // 16 MB
    float*  agg2 = (float*)(ws + 67200000);      // 16 MB (becomes x2 in-place)
    float*  dis  = (float*)(ws + 83200000);      // 0.4 MB (deg then rsqrt in-place)
    __bf16* w1s  = (__bf16*)(ws + 83600000);     // 64 KB  (fragment-swizzled)
    __bf16* w2s  = (__bf16*)(ws + 83665536);     // 6 KB   (fragment-swizzled, padded)
    int*    hist = (int*)(ws + 83671680);        // 256 KB
    float*  gbuf = (float*)(ws + 83933824);      // 16 KB
    int*    misc = (int*)(ws + 83950208);        // 64 B: [0]cnt [1]bin [2]rank [3]thresh [4]loss
    float*  miscf = (float*)misc;
    float*  out  = (float*)d_out;

    // zero accumulators
    zero_u32_kernel<<<(NN + 255) / 256, 256, 0, stream>>>((unsigned*)dis, NN);
    zero_u32_kernel<<<(NN * HID + 255) / 256, 256, 0, stream>>>((unsigned*)agg1, (long long)NN * HID);
    zero_u32_kernel<<<(NN * NC + 255) / 256, 256, 0, stream>>>((unsigned*)agg2, (long long)NN * NC);
    zero_u32_kernel<<<(NBINS + 255) / 256, 256, 0, stream>>>((unsigned*)hist, NBINS);
    zero_u32_kernel<<<1, 256, 0, stream>>>((unsigned*)misc, 16);

    // degree / normalization
    deg_kernel<<<EE / 256, 256, 0, stream>>>(dst, dis);
    dis_kernel<<<(NN + 255) / 256, 256, 0, stream>>>(dis);

    // weights to fragment-swizzled bf16
    conv_w1_kernel<<<(FIN * HID + 255) / 256, 256, 0, stream>>>(w1, w1s);
    conv_w2_kernel<<<(2 * 3 * 32 * 16 + 255) / 256, 256, 0, stream>>>(w2, w2s);

    // layer 1
    gemm1_kernel<<<NN / 32, 64, 0, stream>>>(x, w1s, h0);
    prop1_kernel<<<EE / 8, 256, 0, stream>>>(src, dst, dis, h0, agg1);
    selfloop1_kernel<<<(NN * HID) / 256, 256, 0, stream>>>(h0, agg1, dis, b1);

    // layer 2 (h1 == agg1)
    gemm2_kernel<<<NN / 32, 64, 0, stream>>>(agg1, w2s, g2);
    prop2_kernel<<<EE / 8, 256, 0, stream>>>(src, dst, dis, g2, agg2);
    selfloop2_kernel<<<(NN * NC) / 256, 256, 0, stream>>>(g2, agg2, dis, b2);

    // exact threshold (rank-1600000 order statistic)
    hist_kernel<<<EE / 256, 256, 0, stream>>>(ew, hist);
    scan_select_kernel<<<1, 32, 0, stream>>>(hist, misc);
    gather_kernel<<<EE / 256, 256, 0, stream>>>(ew, misc, gbuf);
    final_select_kernel<<<1, 32, 0, stream>>>(misc, gbuf);

    // loss over edges (x2 == agg2)
    loss_kernel<<<EE / 256, 256, 0, stream>>>(src, dst, agg2, ew, lu, miscf, &miscf[4]);

    // outputs
    logsoftmax_kernel<<<(NN + 255) / 256, 256, 0, stream>>>(agg2, out);
    finalize_kernel<<<1, 32, 0, stream>>>(miscf, out);
}
